// GAT_10471130267749
// MI455X (gfx1250) — matrix operand order repaired and checked by reading, not compiled, not run on hardware
//
#include <hip/hip_runtime.h>
#include <math.h>

#define NEG_SLOPE 0.2f
#define EPS_F 1e-16f

typedef __attribute__((ext_vector_type(16))) __bf16 v16bf;
typedef __attribute__((ext_vector_type(8)))  __bf16 v8bf;
typedef __attribute__((ext_vector_type(8)))  float  v8f;
typedef __attribute__((ext_vector_type(4)))  float  v4f;

// ---------------------------------------------------------------------------
// Order-preserving float<->uint key for atomicMax-based segment max.
// key(f) monotone in f; key==0 decodes to -NaN (non-finite) => treated as 0
// downstream, matching jnp.where(isfinite(m), m, 0).
// ---------------------------------------------------------------------------
__device__ __forceinline__ unsigned fkey(float f) {
  unsigned u = __float_as_uint(f);
  return (u & 0x80000000u) ? ~u : (u | 0x80000000u);
}
__device__ __forceinline__ float fdecode(unsigned k) {
  unsigned u = (k & 0x80000000u) ? (k ^ 0x80000000u) : ~k;
  return __uint_as_float(u);
}

// ---------------------------------------------------------------------------
// WMMA GEMM: C[M,N] = A[M,K] x B[K,N], B supplied transposed as bf16 Bt[N,K].
// One wave -> 16 rows x (NTILES*16) cols. K multiple of 32, M multiple of 16.
// A is fp32 (converted to bf16 in-register) or already bf16.
// ---------------------------------------------------------------------------
template<int NTILES, int KDIM, bool A_F32>
__global__ __launch_bounds__(256)
void gemm_wmma_bf16(const void* __restrict__ Aptr,
                    const __bf16* __restrict__ Bt,
                    float* __restrict__ C, int Mrows) {
  const int lane = (int)(threadIdx.x & 31);
  const int wave = (int)((blockIdx.x * blockDim.x + threadIdx.x) >> 5);
  const int m0 = wave * 16;
  if (m0 >= Mrows) return;                 // wave-uniform: EXEC stays all-ones
  const int l16 = lane & 15;
  const int hi  = lane >> 4;
  const int N   = NTILES * 16;

  v8f acc[NTILES];
#pragma unroll
  for (int t = 0; t < NTILES; ++t)
#pragma unroll
    for (int j = 0; j < 8; ++j) acc[t][j] = 0.f;

  const int row = m0 + l16;
  for (int kk = 0; kk < KDIM; kk += 32) {
    // A fragment: 16-bit A 16x32 layout. Lane holds K = kk+hi*16 .. +15
    // (contiguous, matching B) for row M = l16.
    v16bf a;
    if (A_F32) {
      const float* A = (const float*)Aptr;
      const float* p0 = A + (size_t)row * KDIM + kk + hi * 16;
      v4f x0 = *(const v4f*)(p0);
      v4f x1 = *(const v4f*)(p0 + 4);
      v4f y0 = *(const v4f*)(p0 + 8);
      v4f y1 = *(const v4f*)(p0 + 12);
#pragma unroll
      for (int j = 0; j < 4; ++j) {
        a[j]      = (__bf16)x0[j];
        a[4 + j]  = (__bf16)x1[j];
        a[8 + j]  = (__bf16)y0[j];
        a[12 + j] = (__bf16)y1[j];
      }
    } else {
      const __bf16* A = (const __bf16*)Aptr;
      const __bf16* p0 = A + (size_t)row * KDIM + kk + hi * 16;
      v8bf x = *(const v8bf*)(p0);
      v8bf y = *(const v8bf*)(p0 + 8);
#pragma unroll
      for (int j = 0; j < 8; ++j) { a[j] = x[j]; a[8 + j] = y[j]; }
    }
#pragma unroll
    for (int t = 0; t < NTILES; ++t) {
      // B fragment: lanes 0-15 hold K=kk..kk+15 (hi=0), lanes 16-31 K=kk+16..
      // for col N = t*16 + l16; contiguous in Bt[N][K].
      const __bf16* bp = Bt + (size_t)(t * 16 + l16) * KDIM + kk + hi * 16;
      v16bf b = *(const v16bf*)bp;
      acc[t] = __builtin_amdgcn_wmma_f32_16x16x32_bf16(
          false, a, false, b, (short)0, acc[t], false, false);
    }
  }

  // D layout: VGPR v, lanes 0-15 -> M=m0+v, lanes 16-31 -> M=m0+8+v; N=l16.
  const int orow = m0 + hi * 8;
#pragma unroll
  for (int t = 0; t < NTILES; ++t)
#pragma unroll
    for (int v = 0; v < 8; ++v)
      C[(size_t)(orow + v) * N + t * 16 + l16] = acc[t][v];
}

// ---------------------------------------------------------------------------
// Prep: transpose fp32 W[K,N] -> bf16 Wt[N,K]
// ---------------------------------------------------------------------------
__global__ void prep_wt(const float* __restrict__ W, __bf16* __restrict__ Wt,
                        int K, int N) {
  int i = blockIdx.x * blockDim.x + threadIdx.x;
  if (i >= K * N) return;
  int k = i / N, n = i - k * N;
  Wt[(size_t)n * K + k] = (__bf16)W[i];
}

// ---------------------------------------------------------------------------
// Per-node attention components: alpha[n,h] = sum_c h[n,h*C+c] * att[h,c]
// ---------------------------------------------------------------------------
__global__ void alpha1_kernel(const float* __restrict__ h,
                              const float* __restrict__ a_src,
                              const float* __restrict__ a_dst,
                              float* __restrict__ Asrc, float* __restrict__ Adst,
                              int n_nodes) {
  int i = blockIdx.x * blockDim.x + threadIdx.x;        // over n*8
  if (i >= n_nodes * 8) return;
  int n = i >> 3, hd = i & 7;
  const float* hp = h + (size_t)n * 64 + hd * 8;
  float s = 0.f, d = 0.f;
#pragma unroll
  for (int c = 0; c < 8; ++c) {
    float v = hp[c];
    s += v * a_src[hd * 8 + c];
    d += v * a_dst[hd * 8 + c];
  }
  Asrc[i] = s; Adst[i] = d;
}

__global__ void alpha2_kernel(const float* __restrict__ h,
                              const float* __restrict__ a_src,
                              const float* __restrict__ a_dst,
                              float* __restrict__ Asrc, float* __restrict__ Adst,
                              int n_nodes) {
  int n = blockIdx.x * blockDim.x + threadIdx.x;
  if (n >= n_nodes) return;
  const float* hp = h + (size_t)n * 32;
  float s = 0.f, d = 0.f;
#pragma unroll
  for (int c = 0; c < 32; ++c) {
    float v = hp[c];
    s += v * a_src[c];
    d += v * a_dst[c];
  }
  Asrc[n] = s; Adst[n] = d;
}

// ---------------------------------------------------------------------------
// Edge src/dst with implicit self-loops appended after the E real edges.
// ---------------------------------------------------------------------------
__device__ __forceinline__ void edge_sd(const long long* __restrict__ ei,
                                        long long E, long long e,
                                        int& src, int& dst) {
  if (e < E) { src = (int)ei[e]; dst = (int)ei[E + e]; }
  else       { src = dst = (int)(e - E); }
}

__device__ __forceinline__ float lrelu(float x) {
  return x > 0.f ? x : NEG_SLOPE * x;
}

// ---- conv1 (8 heads): thread per (edge, head) -----------------------------
__global__ void edge_max1(const long long* __restrict__ ei, long long E, int n_nodes,
                          const float* __restrict__ As, const float* __restrict__ Ad,
                          unsigned* __restrict__ mkey) {
  long long tid = (long long)blockIdx.x * blockDim.x + threadIdx.x;
  long long tot = (E + n_nodes) * 8;
  if (tid >= tot) return;
  long long e = tid >> 3; int hd = (int)(tid & 7);
  int src, dst; edge_sd(ei, E, e, src, dst);
  float x = lrelu(As[(size_t)src * 8 + hd] + Ad[(size_t)dst * 8 + hd]);
  atomicMax(&mkey[(size_t)dst * 8 + hd], fkey(x));
}

__global__ void edge_sum1(const long long* __restrict__ ei, long long E, int n_nodes,
                          const float* __restrict__ As, const float* __restrict__ Ad,
                          const unsigned* __restrict__ mkey, float* __restrict__ ssum) {
  long long tid = (long long)blockIdx.x * blockDim.x + threadIdx.x;
  long long tot = (E + n_nodes) * 8;
  if (tid >= tot) return;
  long long e = tid >> 3; int hd = (int)(tid & 7);
  int src, dst; edge_sd(ei, E, e, src, dst);
  float x = lrelu(As[(size_t)src * 8 + hd] + Ad[(size_t)dst * 8 + hd]);
  float m = fdecode(mkey[(size_t)dst * 8 + hd]);
  if (!isfinite(m)) m = 0.f;
  atomicAdd(&ssum[(size_t)dst * 8 + hd], expf(x - m));
}

__global__ void edge_aggr1(const long long* __restrict__ ei, long long E, int n_nodes,
                           const float* __restrict__ As, const float* __restrict__ Ad,
                           const unsigned* __restrict__ mkey,
                           const float* __restrict__ ssum,
                           const float* __restrict__ h, float* __restrict__ agg) {
  long long tid = (long long)blockIdx.x * blockDim.x + threadIdx.x;
  long long tot = (E + n_nodes) * 8;
  if (tid >= tot) return;
  long long e = tid >> 3; int hd = (int)(tid & 7);
  int src, dst; edge_sd(ei, E, e, src, dst);
  float x = lrelu(As[(size_t)src * 8 + hd] + Ad[(size_t)dst * 8 + hd]);
  float m = fdecode(mkey[(size_t)dst * 8 + hd]);
  if (!isfinite(m)) m = 0.f;
  float coef = expf(x - m) / (ssum[(size_t)dst * 8 + hd] + EPS_F);
  const float* hp = h + (size_t)src * 64 + hd * 8;
  float* op = agg + (size_t)dst * 64 + hd * 8;
#pragma unroll
  for (int c = 0; c < 8; ++c) atomicAdd(&op[c], hp[c] * coef);
}

// ---- conv2 (1 head, 32 channels) ------------------------------------------
__global__ void edge_max2(const long long* __restrict__ ei, long long E, int n_nodes,
                          const float* __restrict__ As, const float* __restrict__ Ad,
                          unsigned* __restrict__ mkey) {
  long long e = (long long)blockIdx.x * blockDim.x + threadIdx.x;
  if (e >= E + n_nodes) return;
  int src, dst; edge_sd(ei, E, e, src, dst);
  float x = lrelu(As[src] + Ad[dst]);
  atomicMax(&mkey[dst], fkey(x));
}

__global__ void edge_sum2(const long long* __restrict__ ei, long long E, int n_nodes,
                          const float* __restrict__ As, const float* __restrict__ Ad,
                          const unsigned* __restrict__ mkey, float* __restrict__ ssum) {
  long long e = (long long)blockIdx.x * blockDim.x + threadIdx.x;
  if (e >= E + n_nodes) return;
  int src, dst; edge_sd(ei, E, e, src, dst);
  float x = lrelu(As[src] + Ad[dst]);
  float m = fdecode(mkey[dst]);
  if (!isfinite(m)) m = 0.f;
  atomicAdd(&ssum[dst], expf(x - m));
}

__global__ void edge_aggr2(const long long* __restrict__ ei, long long E, int n_nodes,
                           const float* __restrict__ As, const float* __restrict__ Ad,
                           const unsigned* __restrict__ mkey,
                           const float* __restrict__ ssum,
                           const float* __restrict__ h, float* __restrict__ agg) {
  long long tid = (long long)blockIdx.x * blockDim.x + threadIdx.x;  // (edge, quarter)
  long long tot = (E + n_nodes) * 4;
  if (tid >= tot) return;
  long long e = tid >> 2; int q = (int)(tid & 3);
  int src, dst; edge_sd(ei, E, e, src, dst);
  float x = lrelu(As[src] + Ad[dst]);
  float m = fdecode(mkey[dst]);
  if (!isfinite(m)) m = 0.f;
  float coef = expf(x - m) / (ssum[dst] + EPS_F);
  const float* hp = h + (size_t)src * 32 + q * 8;
  float* op = agg + (size_t)dst * 32 + q * 8;
#pragma unroll
  for (int c = 0; c < 8; ++c) atomicAdd(&op[c], hp[c] * coef);
}

// ---------------------------------------------------------------------------
// finalize1: y = elu(agg1 + b1), stored bf16 (only consumer is GEMM2's A).
// ---------------------------------------------------------------------------
__global__ void finalize1_kernel(const float* __restrict__ agg,
                                 const float* __restrict__ bias,
                                 __bf16* __restrict__ out, int total) {
  int i = blockIdx.x * blockDim.x + threadIdx.x;
  if (i >= total) return;
  float v = agg[i] + bias[i & 63];
  v = v > 0.f ? v : (expf(v) - 1.f);
  out[i] = (__bf16)v;
}

// ---------------------------------------------------------------------------
// finalize2: log_softmax(agg2 + b2). One wave32 per node, lane == channel.
// ---------------------------------------------------------------------------
__global__ void logsoftmax_kernel(const float* __restrict__ agg,
                                  const float* __restrict__ bias,
                                  float* __restrict__ out, int n_nodes) {
  int tid = blockIdx.x * blockDim.x + threadIdx.x;
  int n = tid >> 5, lane = tid & 31;
  if (n >= n_nodes) return;
  float v = agg[(size_t)n * 32 + lane] + bias[lane];
  float mx = v;
#pragma unroll
  for (int off = 16; off > 0; off >>= 1) mx = fmaxf(mx, __shfl_xor(mx, off, 32));
  float sm = expf(v - mx);
#pragma unroll
  for (int off = 16; off > 0; off >>= 1) sm += __shfl_xor(sm, off, 32);
  out[(size_t)n * 32 + lane] = v - mx - logf(sm);
}

// ---------------------------------------------------------------------------
static inline int cdiv(long long a, long long b) { return (int)((a + b - 1) / b); }

extern "C" void kernel_launch(void* const* d_in, const int* in_sizes, int n_in,
                              void* d_out, int out_size, void* d_ws, size_t ws_size,
                              hipStream_t stream) {
  const float*     x    = (const float*)d_in[0];
  const long long* ei   = (const long long*)d_in[1];   // int64 [2,E]
  // d_in[2] = edge_weight (unused by reference GATConv)
  const float* W1  = (const float*)d_in[3];
  const float* as1 = (const float*)d_in[4];
  const float* ad1 = (const float*)d_in[5];
  const float* b1  = (const float*)d_in[6];
  const float* W2  = (const float*)d_in[7];
  const float* as2 = (const float*)d_in[8];
  const float* ad2 = (const float*)d_in[9];
  const float* b2  = (const float*)d_in[10];
  float* out = (float*)d_out;

  const int       Nn = in_sizes[0] / 256;   // 100000
  const long long E  = in_sizes[1] / 2;     // 1600000

  // ---- workspace carve (zeroed accumulation region first) ----
  char* w = (char*)d_ws;
  auto carve = [&](size_t bytes) -> void* {
    void* p = (void*)w; w += (bytes + 255) & ~(size_t)255; return p;
  };
  unsigned* m1key = (unsigned*)carve((size_t)Nn * 8 * 4);
  float*    s1    = (float*)   carve((size_t)Nn * 8 * 4);
  float*    agg1  = (float*)   carve((size_t)Nn * 64 * 4);
  unsigned* m2key = (unsigned*)carve((size_t)Nn * 4);
  float*    s2    = (float*)   carve((size_t)Nn * 4);
  float*    agg2  = (float*)   carve((size_t)Nn * 32 * 4);
  size_t zbytes = (size_t)(w - (char*)d_ws);
  __bf16* W1t  = (__bf16*)carve((size_t)256 * 64 * 2);
  __bf16* W2t  = (__bf16*)carve((size_t)64 * 32 * 2);
  float*  h1   = (float*) carve((size_t)Nn * 64 * 4);
  float*  Asrc1= (float*) carve((size_t)Nn * 8 * 4);
  float*  Adst1= (float*) carve((size_t)Nn * 8 * 4);
  __bf16* h1b  = (__bf16*)carve((size_t)Nn * 64 * 2);
  float*  h2   = (float*) carve((size_t)Nn * 32 * 4);
  float*  Asrc2= (float*) carve((size_t)Nn * 4);
  float*  Adst2= (float*) carve((size_t)Nn * 4);

  hipMemsetAsync(d_ws, 0, zbytes, stream);

  // ---- weight prep (tiny) ----
  prep_wt<<<cdiv(256 * 64, 256), 256, 0, stream>>>(W1, W1t, 256, 64);
  prep_wt<<<cdiv(64 * 32, 256), 256, 0, stream>>>(W2, W2t, 64, 32);

  // ---- conv1 ----
  {
    int waves = cdiv(Nn, 16);
    gemm_wmma_bf16<4, 256, true><<<cdiv((long long)waves * 32, 256), 256, 0, stream>>>(
        (const void*)x, W1t, h1, Nn);
  }
  alpha1_kernel<<<cdiv((long long)Nn * 8, 256), 256, 0, stream>>>(h1, as1, ad1,
                                                                  Asrc1, Adst1, Nn);
  long long tot1 = (E + Nn) * 8;
  edge_max1<<<cdiv(tot1, 256), 256, 0, stream>>>(ei, E, Nn, Asrc1, Adst1, m1key);
  edge_sum1<<<cdiv(tot1, 256), 256, 0, stream>>>(ei, E, Nn, Asrc1, Adst1, m1key, s1);
  edge_aggr1<<<cdiv(tot1, 256), 256, 0, stream>>>(ei, E, Nn, Asrc1, Adst1, m1key, s1,
                                                  h1, agg1);
  finalize1_kernel<<<cdiv((long long)Nn * 64, 256), 256, 0, stream>>>(agg1, b1, h1b,
                                                                      Nn * 64);

  // ---- conv2 ----
  {
    int waves = cdiv(Nn, 16);
    gemm_wmma_bf16<2, 64, false><<<cdiv((long long)waves * 32, 256), 256, 0, stream>>>(
        (const void*)h1b, W2t, h2, Nn);
  }
  alpha2_kernel<<<cdiv(Nn, 256), 256, 0, stream>>>(h2, as2, ad2, Asrc2, Adst2, Nn);
  long long tot2 = E + Nn;
  edge_max2<<<cdiv(tot2, 256), 256, 0, stream>>>(ei, E, Nn, Asrc2, Adst2, m2key);
  edge_sum2<<<cdiv(tot2, 256), 256, 0, stream>>>(ei, E, Nn, Asrc2, Adst2, m2key, s2);
  edge_aggr2<<<cdiv(tot2 * 4, 256), 256, 0, stream>>>(ei, E, Nn, Asrc2, Adst2, m2key,
                                                      s2, h2, agg2);

  // ---- log_softmax ----
  logsoftmax_kernel<<<cdiv((long long)Nn * 32, 256), 256, 0, stream>>>(agg2, b2, out,
                                                                       Nn);
}
